// ScaledDotProductAttention_40716289966518
// MI455X (gfx1250) — matrix operand order repaired and checked by reading, not compiled, not run on hardware
//
#include <hip/hip_runtime.h>
#include <hip/hip_bf16.h>

typedef __attribute__((ext_vector_type(16))) _Float16 v16h;
typedef __attribute__((ext_vector_type(8)))  _Float16 v8h;
typedef __attribute__((ext_vector_type(8)))  float    v8f;
typedef __attribute__((ext_vector_type(4)))  unsigned int u32x4;
typedef __attribute__((ext_vector_type(8)))  int      i32x8;
typedef __attribute__((ext_vector_type(4)))  int      i32x4;

#define BH_N   32
#define SEQ    2048
#define DK     64
#define SCALE  0.125f          // 1/sqrt(64)
#define WAVES  4               // waves per block
#define QB     (WAVES * 16)    // queries per block
#define KBLK   32              // keys per iteration
#define ROWP   68              // padded LDS row stride in floats (64 + 4 pad)

__device__ __forceinline__ float rmax16(float x) {
    x = fmaxf(x, __shfl_xor(x, 1, 32));
    x = fmaxf(x, __shfl_xor(x, 2, 32));
    x = fmaxf(x, __shfl_xor(x, 4, 32));
    x = fmaxf(x, __shfl_xor(x, 8, 32));
    return x;
}
__device__ __forceinline__ float rsum16(float x) {
    x += __shfl_xor(x, 1, 32);
    x += __shfl_xor(x, 2, 32);
    x += __shfl_xor(x, 4, 32);
    x += __shfl_xor(x, 8, 32);
    return x;
}

// Issue one TDM copy of a 32x64 f32 tile (contiguous 2048 dwords) into LDS,
// with LDS padding of 4 dwords every 64 dwords (rows land at stride ROWP).
// D# bitfields per CDNA5 ISA ch.8 (group0: count/lds_addr/global_addr/type,
// group1: data_size, pad ctl, tensor/tile dims, dim0 stride).
__device__ __forceinline__ void tdm_load_tile(const float* gsrc, void* ldst) {
    unsigned long long ga = (unsigned long long)(size_t)gsrc;
    unsigned int la = (unsigned int)(size_t)ldst;   // low 32 bits = LDS offset

    u32x4 g0;
    g0[0] = 1u;                                      // count=1, no gather
    g0[1] = la;                                      // lds_addr
    g0[2] = (unsigned int)ga;                        // global_addr[31:0]
    g0[3] = (unsigned int)(ga >> 32) | (2u << 30);   // global_addr[56:32] | type=2

    i32x8 g1;
    g1[0] = (2 << 16)        // data_size = 4 bytes
          | (1 << 20)        // pad_enable
          | (5 << 22)        // pad_interval: 64 dwords
          | (3 << 25);       // pad_amount: 4 dwords
    g1[1] = (int)(2048u << 16);   // tensor_dim0[15:0] = 2048 (bits 63:48 -> [31:16])
    g1[2] = (int)(1u << 16);      // tensor_dim0[31:16]=0 ; tensor_dim1[15:0] = 1
    g1[3] = (int)(2048u << 16);   // tensor_dim1[31:16]=0 ; tile_dim0 = 2048
    g1[4] = 0;                    // tile_dim1 = 0 (1-D tile), tile_dim2 = 0
    g1[5] = 2048;                 // tensor_dim0_stride[31:0]
    g1[6] = 0;
    g1[7] = 0;

    i32x4 z4; z4[0] = 0; z4[1] = 0; z4[2] = 0; z4[3] = 0;
    i32x8 z8;
    #pragma unroll
    for (int i = 0; i < 8; ++i) z8[i] = 0;

    // 6-arg form on this toolchain: (g0, g1, g2, g3, extra group, cpol)
    __builtin_amdgcn_tensor_load_to_lds(g0, g1, z4, z4, z8, 0);
}

__global__ __launch_bounds__(WAVES * 32)
void fa_fwd_kernel(const float* __restrict__ qg,
                   const float* __restrict__ kg,
                   const float* __restrict__ vg,
                   const float* __restrict__ maskg,
                   float* __restrict__ outg)
{
    // Double-buffered K/V f32 tiles (padded rows) + per-wave P-transpose bounce.
    __shared__ alignas(16) float     ktile[2][KBLK * ROWP];
    __shared__ alignas(16) float     vtile[2][KBLK * ROWP];
    __shared__ alignas(16) _Float16  pstage[WAVES][16 * 32];

    const int lane = threadIdx.x & 31;
    const int wave = threadIdx.x >> 5;
    const int bh   = blockIdx.x / (SEQ / QB);
    const int q0   = (blockIdx.x % (SEQ / QB)) * QB + wave * 16;

    const int half = lane >> 4;
    const int ln   = lane & 15;

    const float* kbh = kg + (size_t)bh * SEQ * DK;
    const float* vbh = vg + (size_t)bh * SEQ * DK;

    // ---------------- Q tile -> A operands (f16), d split 0..31 / 32..63
    const int aM  = ln;
    const int klo = half ? 8 : 0;
    v16h a_lo, a_hi;
    {
        const float* qrow = qg + ((size_t)bh * SEQ + (q0 + aM)) * DK;
        #pragma unroll
        for (int j = 0; j < 8; ++j) {
            a_lo[j]     = (_Float16)qrow[klo + j];
            a_lo[j + 8] = (_Float16)qrow[klo + 16 + j];
            a_hi[j]     = (_Float16)qrow[32 + klo + j];
            a_hi[j + 8] = (_Float16)qrow[32 + klo + 16 + j];
        }
    }

    // ---------------- online-softmax state + output accumulators
    float mrow[8], lrow[8];
    v8f   acc[4];
    #pragma unroll
    for (int r = 0; r < 8; ++r) { mrow[r] = -1e30f; lrow[r] = 0.0f; }
    #pragma unroll
    for (int t = 0; t < 4; ++t)
        #pragma unroll
        for (int j = 0; j < 8; ++j) acc[t][j] = 0.0f;

    const float* maskbase = maskg + (size_t)bh * SEQ * SEQ;
    _Float16* __restrict__ ps = pstage[wave];

    // ---------------- prologue: prefetch first two K/V tiles via TDM
    if (wave == 0) {
        tdm_load_tile(kbh,             &ktile[0][0]);
        tdm_load_tile(vbh,             &vtile[0][0]);
        tdm_load_tile(kbh + KBLK * DK, &ktile[1][0]);
        tdm_load_tile(vbh + KBLK * DK, &vtile[1][0]);
    }

    for (int kb = 0; kb < SEQ; kb += KBLK) {
        const int buf = (kb / KBLK) & 1;

        // Oldest 2 TDM ops (this buffer's K and V) must have landed.
        if (wave == 0) __builtin_amdgcn_s_wait_tensorcnt(2);
        __syncthreads();

        const float* kt = &ktile[buf][0];
        const float* vt = &vtile[buf][0] + ln;

        // ---------- B operands for Q*K^T from LDS: B[d][n] = K[kb+n][d], n=lane%16, d in table order
        v16h bk0_lo, bk0_hi, bk1_lo, bk1_hi;
        #pragma unroll
        for (int n = 0; n < 16; ++n) {
            bk0_lo[n] = (_Float16)kt[ln * ROWP + klo + n + (n & 8)];
            bk0_hi[n] = (_Float16)kt[ln * ROWP + 32 + klo + n + (n & 8)];
            bk1_lo[n] = (_Float16)kt[(ln + 16) * ROWP + klo + n + (n & 8)];
            bk1_hi[n] = (_Float16)kt[(ln + 16) * ROWP + 32 + klo + n + (n & 8)];
        }

        // ---------- scores (two 16x16 f32 tiles), accumulate both d-halves
        v8f zero8;
        #pragma unroll
        for (int j = 0; j < 8; ++j) zero8[j] = 0.0f;
        v8f s0 = __builtin_amdgcn_wmma_f32_16x16x32_f16(false, a_lo, false, bk0_lo,
                                                        (short)0, zero8, false, false);
        s0     = __builtin_amdgcn_wmma_f32_16x16x32_f16(false, a_hi, false, bk0_hi,
                                                        (short)0, s0,    false, false);
        v8f s1 = __builtin_amdgcn_wmma_f32_16x16x32_f16(false, a_lo, false, bk1_lo,
                                                        (short)0, zero8, false, false);
        s1     = __builtin_amdgcn_wmma_f32_16x16x32_f16(false, a_hi, false, bk1_hi,
                                                        (short)0, s1,    false, false);

        // ---------- multiplicative mask (before softmax) + scale
        // Mask is streamed once with zero reuse -> non-temporal loads.
        #pragma unroll
        for (int r = 0; r < 8; ++r) {
            const int m = r + 8 * half;
            const float* mr = maskbase + (size_t)(q0 + m) * SEQ + kb + ln;
            const float mk0 = __builtin_nontemporal_load(mr);
            const float mk1 = __builtin_nontemporal_load(mr + 16);
            s0[r] = s0[r] * SCALE * mk0;
            s1[r] = s1[r] * SCALE * mk1;
        }

        // ---------- online softmax update
        float p0[8], p1[8];
        #pragma unroll
        for (int r = 0; r < 8; ++r) {
            const float tmax = rmax16(fmaxf(s0[r], s1[r]));
            const float nm   = fmaxf(mrow[r], tmax);
            const float corr = __expf(mrow[r] - nm);
            mrow[r] = nm;
            p0[r] = __expf(s0[r] - nm);
            p1[r] = __expf(s1[r] - nm);
            const float rs = rsum16(p0[r] + p1[r]);
            lrow[r] = lrow[r] * corr + rs;
            #pragma unroll
            for (int t = 0; t < 4; ++t) acc[t][r] *= corr;
        }

        // ---------- V tile -> B operands from LDS: B[kk][n] = V[kb+kk][t*16+n]
        v16h bv[4];
        #pragma unroll
        for (int t = 0; t < 4; ++t)
            #pragma unroll
            for (int n = 0; n < 16; ++n)
                bv[t][n] = (_Float16)vt[(klo + n + (n & 8)) * ROWP + t * 16];

        // ---------- P: C-layout -> A-layout via per-wave LDS bounce
        #pragma unroll
        for (int r = 0; r < 8; ++r) {
            const int m = r + 8 * half;
            ps[m * 32 + ln]      = (_Float16)p0[r];
            ps[m * 32 + 16 + ln] = (_Float16)p1[r];
        }
        asm volatile("s_wait_dscnt 0x0" ::: "memory");

        v16h ap;
        {
            const v8h plo = *(const v8h*)&ps[aM * 32 + klo];
            const v8h phi = *(const v8h*)&ps[aM * 32 + klo + 16];
            #pragma unroll
            for (int j = 0; j < 8; ++j) { ap[j] = plo[j]; ap[j + 8] = phi[j]; }
        }
        asm volatile("" ::: "memory");

        // ---------- PV accumulate: 4 d-chunks of 16
        #pragma unroll
        for (int t = 0; t < 4; ++t)
            acc[t] = __builtin_amdgcn_wmma_f32_16x16x32_f16(false, ap, false, bv[t],
                                                            (short)0, acc[t], false, false);

        // ---------- all waves done reading this buffer; refill it via TDM
        __syncthreads();
        if (wave == 0 && (kb + 2 * KBLK) < SEQ) {
            tdm_load_tile(kbh + (kb + 2 * KBLK) * DK, &ktile[buf][0]);
            tdm_load_tile(vbh + (kb + 2 * KBLK) * DK, &vtile[buf][0]);
        }
    }

    // ---------------- normalize and store (non-temporal, streamed once)
    #pragma unroll
    for (int r = 0; r < 8; ++r) {
        const int   m    = r + 8 * half;
        const float rden = 1.0f / lrow[r];
        float* orow = outg + ((size_t)bh * SEQ + (q0 + m)) * DK + ln;
        #pragma unroll
        for (int t = 0; t < 4; ++t)
            __builtin_nontemporal_store(acc[t][r] * rden, orow + t * 16);
    }
}

extern "C" void kernel_launch(void* const* d_in, const int* in_sizes, int n_in,
                              void* d_out, int out_size, void* d_ws, size_t ws_size,
                              hipStream_t stream)
{
    const float* q    = (const float*)d_in[0];
    const float* k    = (const float*)d_in[1];
    const float* v    = (const float*)d_in[2];
    const float* mask = (const float*)d_in[3];
    float* out        = (float*)d_out;

    dim3 grid(BH_N * (SEQ / QB));   // 32 * 32 = 1024 blocks
    dim3 block(WAVES * 32);         // 128 threads = 4 waves
    fa_fwd_kernel<<<grid, block, 0, stream>>>(q, k, v, mask, out);
}
